// CrossAttentionMambaBlock_9749575762042
// MI455X (gfx1250) — matrix operand order repaired and checked by reading, not compiled, not run on hardware
//
#include <hip/hip_runtime.h>
#include <hip/hip_bf16.h>

typedef __attribute__((ext_vector_type(16))) __bf16 v16bf;
typedef __attribute__((ext_vector_type(8)))  __bf16 v8bf;
typedef __attribute__((ext_vector_type(8)))  float  v8f;

__device__ __forceinline__ __bf16 f2bf(float f) { return (__bf16)f; }
__device__ __forceinline__ float relu6f(float v) { return fminf(fmaxf(v, 0.f), 6.f); }
__device__ __forceinline__ float sigmf(float v)  { return 1.f / (1.f + __expf(-v)); }

// =====================================================================
// Generic WMMA bf16 GEMM. Workgroup = 4 waves computing a 64(M) x 64(N)
// tile; each wave owns a 16(M) row-tile and 4 16x16 WMMA N-subtiles that
// share the A fragment. For strided-K activations (xsk != 1) the B tile
// is cooperatively staged in LDS (coalesced global reads, bf16 [n][k]
// layout, ds_load_b128 fragment reads). Out-of-range row/col tiles are
// clamped (duplicate compute) and skipped only at store time.
//   acc = scale * sum_k W[m,k]*X[k,n]
//   accMode: 0: Y = ep(acc)   1: Y += ep(acc)   2: Y = ep(Y + acc)
//   ep(v) = act(gain[m]*v + bias[m]);  act: 0 none, 1 relu6, 2 softplus
// Requirements: M % 16 == 0, N % 16 == 0; if accMode != 0 then N % 64 == 0
// and M % 64 == 0. Strides are 32-bit element counts (max index < 2^31).
// =====================================================================
__global__ __launch_bounds__(128) void gemm_wmma_kernel(
    const float* __restrict__ W, int wbs, int wsm, int wsk,
    const float* __restrict__ X, int xbs, int xsk, int xsn,
    float* __restrict__ Y, int ybs, int ysm, int ysn,
    const float* __restrict__ gain, const float* __restrict__ bias,
    int M, int N, int K, int act, int accMode, float scale)
{
  const int tid  = threadIdx.x;
  const int wave = tid >> 5;
  const int lane = tid & 31;
  const int half = lane >> 4;
  const int mr   = lane & 15;
  const int rowBase0 = blockIdx.y * 64;
  const int colBase0 = blockIdx.x * 64;
  const float* Wb = W + (long)blockIdx.z * wbs;
  const float* Xb = X + (long)blockIdx.z * xbs;
  float*       Yb = Y + (long)blockIdx.z * ybs;

  __shared__ alignas(16) __bf16 Bs[64][40];   // [n][k], 80B rows (16B aligned)

  v8f acc[4];
#pragma unroll
  for (int t = 0; t < 4; ++t)
#pragma unroll
    for (int r = 0; r < 8; ++r) acc[t][r] = 0.f;

  // clamped subtile columns (uniform per wave; duplicates are not stored)
  int colC[4], bOff[4];
#pragma unroll
  for (int t = 0; t < 4; ++t) {
    int col = colBase0 + t * 16;
    colC[t] = (col < N) ? col : (N - 16);
    bOff[t] = (colC[t] + mr) * xsn;                  // B: lane = N col
  }

  const int rowB    = rowBase0 + wave * 16;          // this wave's row tile
  const int rowT    = (rowB < M) ? rowB : (M - 16);  // clamped
  const int arow    = rowT + mr;                     // A: lane = M row
  const int aRowOff = arow * wsm;
  const int Kmain   = K & ~31;

  // cooperative-fill coordinates (strided-B path)
  const int fn  = tid & 63;                          // column within tile
  const int fkb = (tid >> 6) * 16;                   // k base: 0 or 16
  int fcol = colBase0 + fn; if (fcol >= N) fcol = N - 1;
  const float* fxp = Xb + fcol * xsn;

  for (int k0 = 0; k0 < Kmain; k0 += 32) {
    // ---- A fragment: 16x32 bf16, contiguous k = 16*half + e ----
    v16bf a;
    if (wsk == 1) {                                  // contiguous K: 4x b128 loads
      const float* ap = Wb + aRowOff + k0 + half * 16;
      float4 a0 = *(const float4*)(ap);
      float4 a1 = *(const float4*)(ap + 4);
      float4 a2 = *(const float4*)(ap + 8);
      float4 a3 = *(const float4*)(ap + 12);
      __builtin_prefetch(ap + 32, 0, 1);
      a[0] = f2bf(a0.x); a[1] = f2bf(a0.y); a[2] = f2bf(a0.z); a[3] = f2bf(a0.w);
      a[4] = f2bf(a1.x); a[5] = f2bf(a1.y); a[6] = f2bf(a1.z); a[7] = f2bf(a1.w);
      a[8] = f2bf(a2.x); a[9] = f2bf(a2.y); a[10] = f2bf(a2.z); a[11] = f2bf(a2.w);
      a[12] = f2bf(a3.x); a[13] = f2bf(a3.y); a[14] = f2bf(a3.z); a[15] = f2bf(a3.w);
    } else {                                         // strided K: incremental offsets
      const float* ap = Wb + aRowOff;
      int o0 = (k0 + half * 16) * wsk;
      int o1 = (k0 + half * 16 + 8) * wsk;
#pragma unroll
      for (int e = 0; e < 8; ++e) {
        a[e]     = f2bf(ap[o0]);
        a[8 + e] = f2bf(ap[o1]);
        o0 += wsk; o1 += wsk;
      }
    }

    if (xsk == 1) {
      // ---- contiguous-K B: direct 4x b128 loads per subtile ----
#pragma unroll
      for (int t = 0; t < 4; ++t) {
        const float* bp = Xb + bOff[t] + k0 + half * 16;
        float4 b0 = *(const float4*)(bp);
        float4 b1 = *(const float4*)(bp + 4);
        float4 b2 = *(const float4*)(bp + 8);
        float4 b3 = *(const float4*)(bp + 12);
        __builtin_prefetch(bp + 32, 0, 1);
        v16bf b;
        b[0] = f2bf(b0.x); b[1] = f2bf(b0.y); b[2] = f2bf(b0.z); b[3] = f2bf(b0.w);
        b[4] = f2bf(b1.x); b[5] = f2bf(b1.y); b[6] = f2bf(b1.z); b[7] = f2bf(b1.w);
        b[8] = f2bf(b2.x); b[9] = f2bf(b2.y); b[10] = f2bf(b2.z); b[11] = f2bf(b2.w);
        b[12] = f2bf(b3.x); b[13] = f2bf(b3.y); b[14] = f2bf(b3.z); b[15] = f2bf(b3.w);
        acc[t] = __builtin_amdgcn_wmma_f32_16x16x32_bf16(false, a, false, b, (short)0,
                                                         acc[t], false, false);
      }
    } else {
      // ---- strided-K B: cooperative LDS staging, shared by all 4 waves ----
      __syncthreads();                               // previous reads done
      {
        int o = (k0 + fkb) * xsk;
#pragma unroll
        for (int i = 0; i < 8; ++i) {                // 16 k's per thread, packed x2
          int k = fkb + 2 * i;
          unsigned lo = (unsigned)__builtin_bit_cast(unsigned short, f2bf(fxp[o]));
          unsigned hi = (unsigned)__builtin_bit_cast(unsigned short, f2bf(fxp[o + xsk]));
          *(unsigned*)&Bs[fn][k] = lo | (hi << 16);
          o += 2 * xsk;
        }
      }
      __syncthreads();
#pragma unroll
      for (int t = 0; t < 4; ++t) {
        const __bf16* bp = &Bs[colC[t] + mr - colBase0][half * 16];
        v8bf lo = *(const v8bf*)(bp);                // ds_load_b128
        v8bf hi = *(const v8bf*)(bp + 8);            // ds_load_b128
        v16bf b;
#pragma unroll
        for (int e = 0; e < 8; ++e) { b[e] = lo[e]; b[8 + e] = hi[e]; }
        acc[t] = __builtin_amdgcn_wmma_f32_16x16x32_bf16(false, a, false, b, (short)0,
                                                         acc[t], false, false);
      }
    }
  }

  // ---- K tail (K % 32), branch-free: clamped address + select ----
  if (Kmain < K) {
    const int k0 = Kmain;
    v16bf a;
#pragma unroll
    for (int e = 0; e < 16; ++e) {
      int kk = half * 16 + e;
      int kg = k0 + kk;
      int kc = kg < K ? kg : 0;
      float av = Wb[aRowOff + kc * wsk];
      a[e] = f2bf(kg < K ? av : 0.f);
    }
#pragma unroll
    for (int t = 0; t < 4; ++t) {
      v16bf b;
#pragma unroll
      for (int e = 0; e < 16; ++e) {
        int kg = k0 + half * 16 + e;
        int kc = kg < K ? kg : 0;
        float bv = Xb[kc * xsk + bOff[t]];
        b[e] = f2bf(kg < K ? bv : 0.f);
      }
      acc[t] = __builtin_amdgcn_wmma_f32_16x16x32_bf16(false, a, false, b, (short)0,
                                                       acc[t], false, false);
    }
  }

  // ---- epilogue: D f32 layout: vgpr r -> M = r + 8*half, N = lane&15 ----
  if (rowB < M) {
#pragma unroll
    for (int t = 0; t < 4; ++t) {
      int col = colBase0 + t * 16;
      if (col < N) {
        int n = col + mr;
#pragma unroll
        for (int r = 0; r < 8; ++r) {
          int m = rowT + r + half * 8;
          int yi = m * ysm + n * ysn;
          float v = acc[t][r] * scale;
          if (accMode == 2) v += Yb[yi];
          float g = gain ? gain[m] : 1.f;
          float c = bias ? bias[m] : 0.f;
          v = g * v + c;
          if (act == 1)      v = relu6f(v);
          else if (act == 2) v = (v > 20.f) ? v : log1pf(__expf(v));
          if (accMode == 1) v += Yb[yi];
          Yb[yi] = v;
        }
      }
    }
  }
}

// ---------------- elementwise / support kernels ----------------
__global__ void affine_nchw_kernel(const float* __restrict__ x, const float* __restrict__ g,
                                   const float* __restrict__ b, float* __restrict__ y,
                                   int C, int HW, long total) {
  long i = (long)blockIdx.x * blockDim.x + threadIdx.x;
  if (i >= total) return;
  int c = (int)((i / HW) % C);
  y[i] = x[i] * g[c] + b[c];
}

// depthwise KxK conv (NCHW, square, dilated, zero pad) + BN affine + relu6
__global__ void dwconv2d_kernel(const float* __restrict__ in, const float* __restrict__ w,
                                const float* __restrict__ g, const float* __restrict__ b,
                                float* __restrict__ out, int B, int CH, int H, int Wd,
                                int ks, int dil) {
  long total = (long)B * CH * H * Wd;
  long i = (long)blockIdx.x * blockDim.x + threadIdx.x;
  if (i >= total) return;
  int x = (int)(i % Wd);
  int y = (int)((i / Wd) % H);
  int c = (int)((i / ((long)Wd * H)) % CH);
  int pad = dil * (ks - 1) / 2;
  const float* ib = in + (i / ((long)Wd * H)) * H * Wd;
  const float* wc = w + (long)c * ks * ks;
  float s = 0.f;
  for (int ky = 0; ky < ks; ++ky) {
    int yy = y - pad + ky * dil;
    if (yy < 0 || yy >= H) continue;
    for (int kx = 0; kx < ks; ++kx) {
      int xx = x - pad + kx * dil;
      if (xx < 0 || xx >= Wd) continue;
      s += wc[ky * ks + kx] * ib[yy * Wd + xx];
    }
  }
  out[i] = relu6f(s * g[c] + b[c]);
}

// pooled[b,c,16,16] = mean_{4x4}(gate * xn)
__global__ void gate_pool_kernel(const float* __restrict__ gate, const float* __restrict__ xn,
                                 float* __restrict__ pooled, long total) {
  long i = (long)blockIdx.x * blockDim.x + threadIdx.x;
  if (i >= total) return;
  int px = (int)(i % 16);
  int py = (int)((i / 16) % 16);
  long bc = i / 256;
  const float* gb = gate + bc * 4096;
  const float* xb = xn + bc * 4096;
  float s = 0.f;
  for (int dy = 0; dy < 4; ++dy)
    for (int dx = 0; dx < 4; ++dx) {
      int p = (py * 4 + dy) * 64 + (px * 4 + dx);
      s += gb[p] * xb[p];
    }
  pooled[i] = s * (1.f / 16.f);
}

// softmax over rows of 256 (one wave32 per row; 8 elems/lane)
__global__ void softmax256_kernel(float* __restrict__ d, long rows) {
  int wid = threadIdx.x >> 5, lane = threadIdx.x & 31;
  long row = (long)blockIdx.x * 8 + wid;
  if (row >= rows) return;
  float* p = d + row * 256;
  float v[8], mx = -3.0e38f;
#pragma unroll
  for (int e = 0; e < 8; ++e) { v[e] = p[e * 32 + lane]; mx = fmaxf(mx, v[e]); }
  for (int off = 16; off > 0; off >>= 1) mx = fmaxf(mx, __shfl_xor(mx, off, 32));
  float s = 0.f;
#pragma unroll
  for (int e = 0; e < 8; ++e) { v[e] = __expf(v[e] - mx); s += v[e]; }
  for (int off = 16; off > 0; off >>= 1) s += __shfl_xor(s, off, 32);
  float inv = 1.f / s;
#pragma unroll
  for (int e = 0; e < 8; ++e) p[e * 32 + lane] = v[e] * inv;
}

// global average pool over HW per (b,c)  (block=256, LDS reduce)
__global__ void gap_kernel(const float* __restrict__ xn, float* __restrict__ out, int HW) {
  __shared__ float sm[256];
  const float* p = xn + (long)blockIdx.x * HW;
  float s = 0.f;
  for (int i = threadIdx.x; i < HW; i += 256) s += p[i];
  sm[threadIdx.x] = s;
  __syncthreads();
  for (int off = 128; off > 0; off >>= 1) {
    if (threadIdx.x < off) sm[threadIdx.x] += sm[threadIdx.x + off];
    __syncthreads();
  }
  if (threadIdx.x == 0) out[blockIdx.x] = sm[0] / (float)HW;
}

__global__ void se_fc1_kernel(const float* __restrict__ w, const float* __restrict__ bs,
                              const float* __restrict__ in, float* __restrict__ out, int total) {
  int i = blockIdx.x * blockDim.x + threadIdx.x;
  if (i >= total) return;
  int j = i % 64, b = i / 64;
  float s = bs[j];
  for (int c = 0; c < 256; ++c) s += w[j * 256 + c] * in[b * 256 + c];
  out[i] = relu6f(s);
}
__global__ void se_fc2_kernel(const float* __restrict__ w, const float* __restrict__ bs,
                              const float* __restrict__ in, float* __restrict__ out, int total) {
  int i = blockIdx.x * blockDim.x + threadIdx.x;
  if (i >= total) return;
  int c = i % 256, b = i / 256;
  float s = bs[c];
  for (int j = 0; j < 64; ++j) s += w[c * 64 + j] * in[b * 64 + j];
  out[i] = sigmf(s);
}
__global__ void add_se_kernel(float* __restrict__ attn, const float* __restrict__ ca,
                              const float* __restrict__ xn, int C, int HW, long total) {
  long i = (long)blockIdx.x * blockDim.x + threadIdx.x;
  if (i >= total) return;
  long bc = i / HW;
  attn[i] += ca[bc] * xn[i];
}

// layernorm over C=256: one block per (b,l), reads NCHW x, writes (B,L,C)
__global__ void layernorm_kernel(const float* __restrict__ x, const float* __restrict__ g,
                                 const float* __restrict__ b, float* __restrict__ xm,
                                 int C, int HW) {
  __shared__ float sm[256];
  long row = blockIdx.x;              // b*HW + l
  long bb = row / HW, l = row % HW;
  int c = threadIdx.x;
  float v = x[((long)bb * C + c) * HW + l];
  sm[c] = v;
  __syncthreads();
  for (int off = 128; off > 0; off >>= 1) {
    if (c < off) sm[c] += sm[c + off];
    __syncthreads();
  }
  float mean = sm[0] / (float)C;
  __syncthreads();
  float dvv = v - mean;
  sm[c] = dvv * dvv;
  __syncthreads();
  for (int off = 128; off > 0; off >>= 1) {
    if (c < off) sm[c] += sm[c + off];
    __syncthreads();
  }
  float var = sm[0] / (float)C;
  xm[row * C + c] = dvv * rsqrtf(var + 1e-5f) * g[c] + b[c];
}

// mamba causal depthwise conv1d (k=3) + silu: xz(B,L,1024)[:512] -> xa(B,L,512)
__global__ void mamba_conv1d_kernel(const float* __restrict__ xz, const float* __restrict__ cw,
                                    const float* __restrict__ cb, float* __restrict__ xa,
                                    int L, long total) {
  long i = (long)blockIdx.x * blockDim.x + threadIdx.x;
  if (i >= total) return;
  int ci = (int)(i % 512);
  long bl = i / 512;
  long b = bl / L, l = bl % L;
  const float* base = xz + b * (long)L * 1024;
  float s = cb[ci];
#pragma unroll
  for (int j = 0; j < 3; ++j) {
    long ll = l - 2 + j;
    if (ll >= 0) s += cw[ci * 3 + j] * base[ll * 1024 + ci];
  }
  xa[i] = s * sigmf(s);
}

// selective scan: thread per (b, channel), 16-state recurrence in registers
__global__ void mamba_scan_kernel(const float* __restrict__ xa, const float* __restrict__ dt,
                                  const float* __restrict__ dbl, const float* __restrict__ A_log,
                                  float* __restrict__ ys, int B, int L) {
  int t = blockIdx.x * blockDim.x + threadIdx.x;
  if (t >= B * 512) return;
  int ci = t % 512, b = t / 512;
  float negA[16], h[16];
#pragma unroll
  for (int s = 0; s < 16; ++s) { negA[s] = -__expf(A_log[ci * 16 + s]); h[s] = 0.f; }
  const float* xab = xa + (long)b * L * 512;
  const float* dtb = dt + (long)b * L * 512;
  const float* dbb = dbl + (long)b * L * 48;
  float* yb = ys + (long)b * L * 512;
  for (int l = 0; l < L; ++l) {
    float xv = xab[(long)l * 512 + ci];
    float dv = dtb[(long)l * 512 + ci];
    const float* dr = dbb + (long)l * 48;
    float acc = 0.f;
#pragma unroll
    for (int s = 0; s < 16; ++s) {
      h[s] = h[s] * __expf(dv * negA[s]) + dv * dr[16 + s] * xv;
      acc += h[s] * dr[32 + s];
    }
    yb[(long)l * 512 + ci] = acc;
  }
}

// y = (ys + xa*D[ci]) * silu(z)
__global__ void ygate_kernel(float* __restrict__ ys, const float* __restrict__ xa,
                             const float* __restrict__ D, const float* __restrict__ xz,
                             int L, long total) {
  long i = (long)blockIdx.x * blockDim.x + threadIdx.x;
  if (i >= total) return;
  int ci = (int)(i % 512);
  long bl = i / 512;
  float z = xz[bl * 1024 + 512 + ci];
  ys[i] = (ys[i] + xa[i] * D[ci]) * (z * sigmf(z));
}

__global__ void combine_kernel(const float* __restrict__ x, const float* __restrict__ attn,
                               const float* __restrict__ fused, const float* __restrict__ alpha,
                               float* __restrict__ out, long total) {
  long i = (long)blockIdx.x * blockDim.x + threadIdx.x;
  if (i >= total) return;
  float a = alpha[0];
  out[i] = x[i] + a * attn[i] + (1.f - a) * fused[i];
}

__global__ void add_inplace_kernel(float* __restrict__ a, const float* __restrict__ b, long total) {
  long i = (long)blockIdx.x * blockDim.x + threadIdx.x;
  if (i >= total) return;
  a[i] += b[i];
}

__global__ void zero_kernel(float* __restrict__ p, long total) {
  long i = (long)blockIdx.x * blockDim.x + threadIdx.x;
  if (i < total) p[i] = 0.f;
}

// =====================================================================
static inline void gemm(hipStream_t s,
                        const float* W, int wbs, int wsm, int wsk,
                        const float* X, int xbs, int xsk, int xsn,
                        float* Y, int ybs, int ysm, int ysn,
                        const float* g, const float* b,
                        int M, int N, int K, int act, int accMode, float scale, int batches) {
  dim3 gr((N + 63) / 64, (M + 63) / 64, batches);
  gemm_wmma_kernel<<<gr, dim3(128), 0, s>>>(W, wbs, wsm, wsk, X, xbs, xsk, xsn,
                                            Y, ybs, ysm, ysn, g, b, M, N, K, act, accMode, scale);
}
static inline dim3 eb(long total) { return dim3((unsigned)((total + 255) / 256)); }

extern "C" void kernel_launch(void* const* d_in, const int* in_sizes, int n_in,
                              void* d_out, int out_size, void* d_ws, size_t ws_size,
                              hipStream_t stream) {
  (void)in_sizes; (void)n_in;
  const int B = 2, C = 256, H = 64, Wd = 64, HW = 4096, CQ = 64, L = 4096, DI = 512;
  const long TOT = (long)B * C * HW;   // 2,097,152

  // ---- inputs (setup_inputs dict order, recursively flattened) ----
  const float* x        = (const float*)d_in[0];
  const float* n1_g     = (const float*)d_in[1];
  const float* n1_b     = (const float*)d_in[2];
  const float* w0_1     = (const float*)d_in[3];
  const float* b01g     = (const float*)d_in[4];
  const float* b01b     = (const float*)d_in[5];
  const float* w0_2     = (const float*)d_in[6];
  const float* b02g     = (const float*)d_in[7];
  const float* b02b     = (const float*)d_in[8];
  const float* w0_3     = (const float*)d_in[9];
  const float* b03g     = (const float*)d_in[10];
  const float* b03b     = (const float*)d_in[11];
  const float* w1_2     = (const float*)d_in[12];
  const float* b12g     = (const float*)d_in[13];
  const float* b12b     = (const float*)d_in[14];
  const float* w1_3     = (const float*)d_in[15];
  const float* b13g     = (const float*)d_in[16];
  const float* b13b     = (const float*)d_in[17];
  const float* w2_2     = (const float*)d_in[18];
  const float* b22g     = (const float*)d_in[19];
  const float* b22b     = (const float*)d_in[20];
  const float* w2_3     = (const float*)d_in[21];
  const float* b23g     = (const float*)d_in[22];
  const float* b23b     = (const float*)d_in[23];
  const float* w3       = (const float*)d_in[24];
  const float* b3g      = (const float*)d_in[25];
  const float* b3b      = (const float*)d_in[26];
  const float* kv_w     = (const float*)d_in[27];
  const float* fc1_w    = (const float*)d_in[28];
  const float* fc1_b    = (const float*)d_in[29];
  const float* fc2_w    = (const float*)d_in[30];
  const float* fc2_b    = (const float*)d_in[31];
  const float* ln_g     = (const float*)d_in[32];
  const float* ln_b     = (const float*)d_in[33];
  const float* in_proj  = (const float*)d_in[34];
  const float* conv_w   = (const float*)d_in[35];
  const float* conv_b   = (const float*)d_in[36];
  const float* x_proj   = (const float*)d_in[37];
  const float* dt_proj  = (const float*)d_in[38];
  const float* dt_bias  = (const float*)d_in[39];
  const float* A_log    = (const float*)d_in[40];
  const float* Dp       = (const float*)d_in[41];
  const float* out_proj = (const float*)d_in[42];
  const float* fusion_w = (const float*)d_in[43];
  const float* fbn_g    = (const float*)d_in[44];
  const float* fbn_b    = (const float*)d_in[45];
  const float* mfc1_w   = (const float*)d_in[46];
  const float* mfc1g    = (const float*)d_in[47];
  const float* mfc1b    = (const float*)d_in[48];
  const float* c1_w     = (const float*)d_in[49];
  const float* c1g      = (const float*)d_in[50];
  const float* c1b      = (const float*)d_in[51];
  const float* c2_w     = (const float*)d_in[52];
  const float* c2g      = (const float*)d_in[53];
  const float* c2b      = (const float*)d_in[54];
  const float* mfc2_w   = (const float*)d_in[55];
  const float* mfc2g    = (const float*)d_in[56];
  const float* mfc2b    = (const float*)d_in[57];
  const float* alpha    = (const float*)d_in[58];

  // ---- workspace layout (floats) ----
  float* wsp = (float*)d_ws;
  size_t off = 0;
  auto alloc = [&](size_t n) { float* p = wsp + off; off += n; return p; };
  float* xn      = alloc(TOT);
  float* a01     = alloc((size_t)B * CQ * HW);
  float* dwb     = alloc((size_t)B * CQ * HW);
  float* bsum    = alloc(TOT);
  float* gate    = alloc(TOT);
  float* pool    = alloc((size_t)B * C * 256);
  float* kbuf    = alloc((size_t)B * C * 256);
  float* vbuf    = alloc((size_t)B * C * 256);
  float* dots    = alloc((size_t)16 * 4096 * 256);
  float* attnout = alloc(TOT);
  float* ca0     = alloc((size_t)B * C);
  float* ca1     = alloc((size_t)B * CQ);
  float* ca2     = alloc((size_t)B * C);
  float* xm      = alloc((size_t)B * L * C);
  float* xz      = alloc((size_t)B * L * 1024);
  float* xa      = alloc((size_t)B * L * DI);
  float* dbl     = alloc((size_t)B * L * 48);
  float* dtb     = alloc((size_t)B * L * DI);
  float* ys      = alloc((size_t)B * L * DI);
  float* mbuf    = alloc(TOT);
  float* fused   = alloc(TOT);
  float* xcomb   = alloc(TOT);
  float* t1      = alloc((size_t)B * CQ * HW);
  float* t2      = alloc((size_t)B * CQ * HW);
  float* t3      = alloc((size_t)B * CQ * HW);
  if (off * sizeof(float) > ws_size) {   // deterministic fallback if ws too small
    zero_kernel<<<eb(out_size), 256, 0, stream>>>((float*)d_out, out_size);
    return;
  }

  // ---- 1. norm1: xn = x*g + b ----
  affine_nchw_kernel<<<eb(TOT), 256, 0, stream>>>(x, n1_g, n1_b, xn, C, HW, TOT);

  // ---- 2. MSC multiscale gate ----
  gemm(stream, w0_1, 0, 256, 1, xn, C * HW, HW, 1, a01, CQ * HW, HW, 1,
       b01g, b01b, CQ, HW, C, 1, 0, 1.f, B);
  dwconv2d_kernel<<<eb((long)B * CQ * HW), 256, 0, stream>>>(a01, w0_2, b02g, b02b, dwb, B, CQ, H, Wd, 3, 3);
  gemm(stream, w0_3, 0, 64, 1, dwb, CQ * HW, HW, 1, bsum, C * HW, HW, 1,
       b03g, b03b, C, HW, CQ, 1, 0, 1.f, B);
  dwconv2d_kernel<<<eb((long)B * CQ * HW), 256, 0, stream>>>(a01, w1_2, b12g, b12b, dwb, B, CQ, H, Wd, 3, 5);
  gemm(stream, w1_3, 0, 64, 1, dwb, CQ * HW, HW, 1, bsum, C * HW, HW, 1,
       b13g, b13b, C, HW, CQ, 1, 1, 1.f, B);
  dwconv2d_kernel<<<eb((long)B * CQ * HW), 256, 0, stream>>>(a01, w2_2, b22g, b22b, dwb, B, CQ, H, Wd, 3, 7);
  gemm(stream, w2_3, 0, 64, 1, dwb, CQ * HW, HW, 1, bsum, C * HW, HW, 1,
       b23g, b23b, C, HW, CQ, 1, 1, 1.f, B);
  gemm(stream, w3, 0, 256, 1, bsum, C * HW, HW, 1, gate, C * HW, HW, 1,
       b3g, b3b, C, HW, C, 1, 0, 1.f, B);
  gate_pool_kernel<<<eb((long)B * C * 256), 256, 0, stream>>>(gate, xn, pool, (long)B * C * 256);

  // ---- 3. KV projection (split K / V) ----
  gemm(stream, kv_w, 0, 256, 1, pool, C * 256, 256, 1, kbuf, C * 256, 256, 1,
       nullptr, nullptr, C, 256, C, 0, 0, 1.f, B);
  gemm(stream, kv_w + 256 * 256, 0, 256, 1, pool, C * 256, 256, 1, vbuf, C * 256, 256, 1,
       nullptr, nullptr, C, 256, C, 0, 0, 1.f, B);

  // ---- 4. attention: dots = Q·K^T * d^-0.5, softmax, out = P·V ----
  // z = b*8+h; Q addressed from xn via strides (per-z stride 32*4096 / 32*256)
  gemm(stream, xn, 32 * 4096, 1, 4096, kbuf, 32 * 256, 256, 1,
       dots, 4096 * 256, 256, 1, nullptr, nullptr,
       4096, 256, 32, 0, 0, 0.17677669529663687f, 16);
  softmax256_kernel<<<dim3(8192), 256, 0, stream>>>(dots, (long)16 * 4096);
  gemm(stream, dots, 4096 * 256, 256, 1, vbuf, 32 * 256, 1, 256,
       attnout, 32 * 4096, 1, 4096, nullptr, nullptr,
       4096, 32, 256, 0, 0, 1.f, 16);

  // ---- 5. squeeze-excite channel attention ----
  gap_kernel<<<dim3(B * C), 256, 0, stream>>>(xn, ca0, HW);
  se_fc1_kernel<<<dim3(1), B * CQ, 0, stream>>>(fc1_w, fc1_b, ca0, ca1, B * CQ);
  se_fc2_kernel<<<dim3(2), 256, 0, stream>>>(fc2_w, fc2_b, ca1, ca2, B * C);
  add_se_kernel<<<eb(TOT), 256, 0, stream>>>(attnout, ca2, xn, C, HW, TOT);

  // ---- 6. layernorm of original x -> xm (B,L,C) ----
  layernorm_kernel<<<dim3(B * HW), 256, 0, stream>>>(x, ln_g, ln_b, xm, C, HW);

  // ---- 7. Mamba ----
  gemm(stream, in_proj, 0, 256, 1, xm, L * C, 1, C, xz, L * 1024, 1, 1024,
       nullptr, nullptr, 1024, L, C, 0, 0, 1.f, B);
  mamba_conv1d_kernel<<<eb((long)B * L * DI), 256, 0, stream>>>(xz, conv_w, conv_b, xa, L, (long)B * L * DI);
  gemm(stream, x_proj, 0, 512, 1, xa, L * DI, 1, DI, dbl, L * 48, 1, 48,
       nullptr, nullptr, 48, L, DI, 0, 0, 1.f, B);
  gemm(stream, dt_proj, 0, 16, 1, dbl, L * 48, 1, 48, dtb, L * DI, 1, DI,
       nullptr, dt_bias, DI, L, 16, 2, 0, 1.f, B);
  mamba_scan_kernel<<<dim3(4), 256, 0, stream>>>(xa, dtb, dbl, A_log, ys, B, L);
  ygate_kernel<<<eb((long)B * L * DI), 256, 0, stream>>>(ys, xa, Dp, xz, L, (long)B * L * DI);
  gemm(stream, out_proj, 0, 512, 1, ys, L * DI, 1, DI, mbuf, C * HW, HW, 1,
       nullptr, nullptr, C, HW, DI, 0, 0, 1.f, B);

  // ---- 8. fusion conv over concat [attnout, mbuf] + BN (2-pass) ----
  gemm(stream, fusion_w, 0, 512, 1, attnout, C * HW, HW, 1, fused, C * HW, HW, 1,
       nullptr, nullptr, C, HW, C, 0, 0, 1.f, B);
  gemm(stream, fusion_w + 256, 0, 512, 1, mbuf, C * HW, HW, 1, fused, C * HW, HW, 1,
       fbn_g, fbn_b, C, HW, C, 0, 2, 1.f, B);

  // ---- 9. residual combine ----
  combine_kernel<<<eb(TOT), 256, 0, stream>>>(x, attnout, fused, alpha, xcomb, TOT);

  // ---- 10. FFN, accumulated into d_out on top of xcomb ----
  hipMemcpyAsync(d_out, xcomb, TOT * sizeof(float), hipMemcpyDeviceToDevice, stream);
  gemm(stream, mfc1_w, 0, 256, 1, xcomb, C * HW, HW, 1, t1, CQ * HW, HW, 1,
       mfc1g, mfc1b, CQ, HW, C, 1, 0, 1.f, B);
  dwconv2d_kernel<<<eb((long)B * CQ * HW), 256, 0, stream>>>(t1, c1_w, c1g, c1b, t2, B, CQ, H, Wd, 5, 1);
  dwconv2d_kernel<<<eb((long)B * CQ * HW), 256, 0, stream>>>(t1, c2_w, c2g, c2b, t3, B, CQ, H, Wd, 3, 1);
  add_inplace_kernel<<<eb((long)B * CQ * HW), 256, 0, stream>>>(t2, t3, (long)B * CQ * HW);
  gemm(stream, mfc2_w, 0, 64, 1, t2, CQ * HW, HW, 1, (float*)d_out, C * HW, HW, 1,
       mfc2g, mfc2b, C, HW, CQ, 1, 1, 1.f, B);
}
